// Model_58772332479018
// MI455X (gfx1250) — matrix operand order repaired and checked by reading, not compiled, not run on hardware
//
#include <hip/hip_runtime.h>
#include <math.h>

// ---------------------------------------------------------------------------
// Types for CDNA5 WMMA (wave32). bf16 16x16x32, f32 accumulate.
// ---------------------------------------------------------------------------
typedef __attribute__((ext_vector_type(8)))  __bf16 v8bf;
typedef __attribute__((ext_vector_type(16))) __bf16 v16bf;
typedef __attribute__((ext_vector_type(8)))  float  v8f;
typedef __attribute__((ext_vector_type(4)))  float  v4f;

#define WMMA_BF16(a, b, c) \
  __builtin_amdgcn_wmma_f32_16x16x32_bf16(false, (a), false, (b), (short)0, (c), false, false)

__device__ __forceinline__ v8f zero8() {
  v8f z;
#pragma unroll
  for (int i = 0; i < 8; ++i) z[i] = 0.0f;
  return z;
}

// A-operand fragment (16x32): lane m=lane&15, half=lane>>4 holds
// K = half*8 + [0..7] and 16 + half*8 + [0..7]  -> two 16B LDS loads.
__device__ __forceinline__ v16bf frag_a(const __bf16* p, int ko, int half) {
  const v8bf lo = *(const v8bf*)(p + ko + half * 8);
  const v8bf hi = *(const v8bf*)(p + ko + 16 + half * 8);
  v16bf r;
#pragma unroll
  for (int i = 0; i < 8; ++i) { r[i] = lo[i]; r[8 + i] = hi[i]; }
  return r;
}

// B-operand fragment (32x16): lane n=lane&15, half=lane>>4 holds
// K = half*8 + [0..7] and 16 + half*8 + [0..7]  -> two 16B loads from a K-contiguous row of W[n][k].
__device__ __forceinline__ v16bf frag_b(const __bf16* p, int ko, int half) {
  const v8bf lo = *(const v8bf*)(p + ko + half * 8);
  const v8bf hi = *(const v8bf*)(p + ko + 16 + half * 8);
  v16bf r;
#pragma unroll
  for (int i = 0; i < 8; ++i) { r[i] = lo[i]; r[8 + i] = hi[i]; }
  return r;
}

// ---------------------------------------------------------------------------
// gfx1250 async global->LDS staging (ASYNCcnt path), guarded fallback.
// Builtin signature (from clang diagnostic): arg0 is AS1 pointer to
// int __attribute__((vector_size(16))); arg1 the LDS side; then offset, cpol.
// ---------------------------------------------------------------------------
#if __has_builtin(__builtin_amdgcn_global_load_async_to_lds_b128)
#define HAS_ASYNC_LDS 1
typedef int v4i_vs __attribute__((vector_size(16)));
typedef __attribute__((address_space(1))) v4i_vs* gas_p;
typedef __attribute__((address_space(3))) v4i_vs* las_p;
__device__ __forceinline__ void cp16(const __bf16* g, __bf16* l) {
  __builtin_amdgcn_global_load_async_to_lds_b128(
      (gas_p)(uintptr_t)g,
      (las_p)(unsigned)(uintptr_t)l, 0, 0);
}
__device__ __forceinline__ void async_wait() {
  asm volatile("s_wait_asynccnt 0" ::: "memory");
}
#else
#define HAS_ASYNC_LDS 0
__device__ __forceinline__ void cp16(const __bf16* g, __bf16* l) {
  *(uint4*)l = *(const uint4*)g;
}
__device__ __forceinline__ void async_wait() {}
#endif

// ---------------------------------------------------------------------------
// Elementwise conversions
// ---------------------------------------------------------------------------
__global__ void cvt_bf16(const float* __restrict__ in, __bf16* __restrict__ out, int n) {
  int i = blockIdx.x * 256 + threadIdx.x;
  if (i < n) out[i] = (__bf16)in[i];
}

// w_red (O=512, I=512, T=7) -> bf16 [o][t*512 + i]  (K = t*512 + i, K-contiguous)
__global__ void cvt_wred(const float* __restrict__ in, __bf16* __restrict__ out) {
  int i = blockIdx.x * 256 + threadIdx.x;
  if (i < 512 * 512 * 7) {
    int o = i / (512 * 7);
    int rem = i - o * (512 * 7);
    int ii = rem / 7;
    int tt = rem - ii * 7;
    out[(size_t)o * 3584 + tt * 512 + ii] = (__bf16)in[i];
  }
}

// ---------------------------------------------------------------------------
// LayerNorm over C=512, one wave per token, f32 in -> bf16 out
// ---------------------------------------------------------------------------
__global__ __launch_bounds__(256) void ln_bf16(const float* __restrict__ in,
                                               const float* __restrict__ gamma,
                                               const float* __restrict__ beta,
                                               __bf16* __restrict__ out, int ntok) {
  int w = threadIdx.x >> 5, lane = threadIdx.x & 31;
  int tok = blockIdx.x * 8 + w;
  if (tok >= ntok) return;
  const float* p = in + (size_t)tok * 512;
  float vals[16];
  float s = 0.0f, sq = 0.0f;
#pragma unroll
  for (int ch = 0; ch < 4; ++ch) {
    v4f v = *(const v4f*)(p + ch * 128 + lane * 4);
#pragma unroll
    for (int j = 0; j < 4; ++j) {
      vals[ch * 4 + j] = v[j];
      s += v[j];
      sq += v[j] * v[j];
    }
  }
#pragma unroll
  for (int off = 1; off < 32; off <<= 1) {
    s += __shfl_xor(s, off);
    sq += __shfl_xor(sq, off);
  }
  float mean = s * (1.0f / 512.0f);
  float var = sq * (1.0f / 512.0f) - mean * mean;
  float rstd = rsqrtf(var + 1e-5f);
  __bf16* op = out + (size_t)tok * 512;
#pragma unroll
  for (int ch = 0; ch < 4; ++ch) {
#pragma unroll
    for (int j = 0; j < 4; ++j) {
      int c = ch * 128 + lane * 4 + j;
      op[c] = (__bf16)((vals[ch * 4 + j] - mean) * rstd * gamma[c] + beta[c]);
    }
  }
}

// ---------------------------------------------------------------------------
// Tiled bf16 WMMA GEMM:  C[m][n] = sum_k A[m][k] * B[n][k]  (+ Res if HAS_RES)
// MODE 0: dense A.  MODE 1: reducer im2col gather from h (B=4, L=4096):
//   row m -> (b=m>>10, lr=m&1023); k -> (t=k>>9, i=k&511),
//   pos = reflect(lr*4 + t - 3, 4096); element = h[b][pos][i].
// Block 128x128, K-tile 64, double buffered. 8 waves as 4(M) x 2(N);
// each wave: 32x64 via 2x4 accumulators -> 16 WMMA per K-tile.
// All 12 fragments of a K-tile are preloaded before the WMMA burst so one
// ds wait covers 16 WMMAs.
// ---------------------------------------------------------------------------
template <int MODE, bool HAS_RES>
__global__ __launch_bounds__(256) void gemm_bf16(const __bf16* __restrict__ A,
                                                 const __bf16* __restrict__ Bm,
                                                 const float* __restrict__ Res,
                                                 float* __restrict__ Cout,
                                                 int M, int N, int K) {
  __shared__ __align__(16) __bf16 Asm[2][128][64];
  __shared__ __align__(16) __bf16 Bsm[2][128][64];

  const int bn = blockIdx.x, bm = blockIdx.y;
  const int tid = threadIdx.x;
  const int wv = tid >> 5, lane = tid & 31, lq = lane & 15, half = lane >> 4;
  const int wm = wv >> 1, wn = wv & 1;  // wave tile: rows wm*32..+32, cols wn*64..+64

  const int srow = tid >> 1, scc = (tid & 1) * 32;  // staging: 128 rows x 2 chunks of 32
  const int tiles = K / 64;

  auto stage = [&](int buf, int kt) {
    const int k0 = kt * 64;
    const __bf16* asrc;
    if (MODE == 0) {
      asrc = A + (size_t)(bm * 128 + srow) * K + k0 + scc;
    } else {
      int rowg = bm * 128 + srow;
      int b = rowg >> 10, lr = rowg & 1023;
      int k = k0 + scc;
      int tt = k >> 9, ii = k & 511;
      int pos = lr * 4 + tt - 3;
      pos = (pos < 0) ? -pos : ((pos >= 4096) ? (8190 - pos) : pos);
      asrc = A + ((size_t)(b * 4096 + pos) * 512 + ii);
    }
    __bf16* adst = &Asm[buf][srow][scc];
    cp16(asrc, adst);
    cp16(asrc + 8, adst + 8);
    const __bf16* bsrc = Bm + (size_t)(bn * 128 + srow) * K + k0 + scc;
    __bf16* bdst = &Bsm[buf][srow][scc];
    cp16(bsrc, bdst);
    cp16(bsrc + 8, bdst + 8);
    if (kt + 2 < tiles) {
      if (MODE == 0) __builtin_prefetch(asrc + 128, 0, 1);
      __builtin_prefetch(bsrc + 128, 0, 1);
    }
  };

  v8f acc[2][4];
#pragma unroll
  for (int ms = 0; ms < 2; ++ms)
#pragma unroll
    for (int nt = 0; nt < 4; ++nt) acc[ms][nt] = zero8();

  stage(0, 0);
  async_wait();
  __syncthreads();

  for (int kt = 0; kt < tiles; ++kt) {
    const int buf = kt & 1;
    if (kt + 1 < tiles) stage(buf ^ 1, kt + 1);

    // preload all fragments for this K-tile (2 k-steps x (2 A + 4 B))
    v16bf af[2][2], bfr[2][4];
#pragma unroll
    for (int ks = 0; ks < 2; ++ks) {
      const int ko = ks * 32;
#pragma unroll
      for (int ms = 0; ms < 2; ++ms)
        af[ks][ms] = frag_a(&Asm[buf][wm * 32 + ms * 16 + lq][0], ko, half);
#pragma unroll
      for (int nt = 0; nt < 4; ++nt)
        bfr[ks][nt] = frag_b(&Bsm[buf][wn * 64 + nt * 16 + lq][0], ko, half);
    }
#pragma unroll
    for (int ks = 0; ks < 2; ++ks)
#pragma unroll
      for (int ms = 0; ms < 2; ++ms)
#pragma unroll
        for (int nt = 0; nt < 4; ++nt)
          acc[ms][nt] = WMMA_BF16(af[ks][ms], bfr[ks][nt], acc[ms][nt]);

    async_wait();
    __syncthreads();
  }

  // Epilogue: D layout lane = (n=lq, rows half*8 + r in VGPR r)
#pragma unroll
  for (int ms = 0; ms < 2; ++ms) {
#pragma unroll
    for (int nt = 0; nt < 4; ++nt) {
#pragma unroll
      for (int r = 0; r < 8; ++r) {
        int rowg = bm * 128 + wm * 32 + ms * 16 + half * 8 + r;
        int colg = bn * 128 + wn * 64 + nt * 16 + lq;
        size_t idx = (size_t)rowg * N + colg;
        float v = acc[ms][nt][r];
        if (HAS_RES) v += Res[idx];
        Cout[idx] = v;
      }
    }
  }
}

// ---------------------------------------------------------------------------
// Flash attention (bf16 WMMA, f32 online softmax).
// Q: (B,L,512) bf16  K,V: (B,Lr,512) bf16  channel = h*64 + d.
// grid: (L/128, B*H); 8 waves x 16 q-rows; keys processed 32 at a time.
// ---------------------------------------------------------------------------
__global__ __launch_bounds__(256) void attn_kernel(const __bf16* __restrict__ Q,
                                                   const __bf16* __restrict__ Kb,
                                                   const __bf16* __restrict__ Vb,
                                                   __bf16* __restrict__ Out) {
  const int L = 4096, Lr = 1024, H = 8;
  const int b = blockIdx.y / H, h = blockIdx.y % H;
  const int q0 = blockIdx.x * 128;
  const int tid = threadIdx.x;
  const int w = tid >> 5, lane = tid & 31, lq = lane & 15, half = lane >> 4;

  __shared__ __align__(16) __bf16 Ksm[32][64];    // [key][d]
  __shared__ __align__(16) __bf16 Vt[64][32];     // [d][key] (transposed)
  __shared__ __align__(16) __bf16 Psm[8][16][32]; // per-wave P tile [qrow][key]

  const int qrow = q0 + w * 16 + lq;
  const __bf16* qbase = Q + ((size_t)(b * L + qrow) * 512 + h * 64);
  const v16bf qa0 = frag_a(qbase, 0, half);
  const v16bf qa1 = frag_a(qbase, 32, half);

  float m_run[8], l_run[8];
  v8f O[4];
#pragma unroll
  for (int r = 0; r < 8; ++r) { m_run[r] = -3.0e38f; l_run[r] = 0.0f; }
#pragma unroll
  for (int nt = 0; nt < 4; ++nt) O[nt] = zero8();

  for (int kt = 0; kt < Lr; kt += 32) {
    // stage K tile (32x64), key-major, via async LDS path
    if (tid < 128) {
      int row = tid >> 2, cc = (tid & 3) * 16;
      const __bf16* src = Kb + ((size_t)(b * Lr + kt + row) * 512 + h * 64 + cc);
      __bf16* dst = &Ksm[row][cc];
      cp16(src, dst);
      cp16(src + 8, dst + 8);
    }
    // stage V transposed: Vt[d][key]
    {
      int idx = tid * 8;
      int key = idx >> 6, d0 = idx & 63;
      const __bf16* vs = Vb + ((size_t)(b * Lr + kt + key) * 512 + h * 64 + d0);
#pragma unroll
      for (int j = 0; j < 8; ++j) Vt[d0 + j][key] = vs[j];
    }
    async_wait();
    __syncthreads();

    // preload K fragments for both 16-key subtiles, then 4 WMMAs
    v16bf kb[2][2];
#pragma unroll
    for (int st = 0; st < 2; ++st) {
      kb[st][0] = frag_b(&Ksm[st * 16 + lq][0], 0, half);
      kb[st][1] = frag_b(&Ksm[st * 16 + lq][0], 32, half);
    }
    v8f stile[2];
    float tmax[8];
#pragma unroll
    for (int r = 0; r < 8; ++r) tmax[r] = -3.0e38f;
#pragma unroll
    for (int st = 0; st < 2; ++st) {
      v8f s = zero8();
      s = WMMA_BF16(qa0, kb[st][0], s);
      s = WMMA_BF16(qa1, kb[st][1], s);
#pragma unroll
      for (int r = 0; r < 8; ++r) {
        s[r] *= 0.125f;  // 1/sqrt(64)
        tmax[r] = fmaxf(tmax[r], s[r]);
      }
      stile[st] = s;
    }
#pragma unroll
    for (int off = 1; off < 16; off <<= 1)
#pragma unroll
      for (int r = 0; r < 8; ++r) tmax[r] = fmaxf(tmax[r], __shfl_xor(tmax[r], off));

    float m_new[8], fac[8], psum[8];
#pragma unroll
    for (int r = 0; r < 8; ++r) {
      m_new[r] = fmaxf(m_run[r], tmax[r]);
      fac[r] = __expf(m_run[r] - m_new[r]);
      m_run[r] = m_new[r];
      psum[r] = 0.0f;
    }

    // preload V fragments (independent of Psm writes)
    v16bf vfr[4];
#pragma unroll
    for (int nt = 0; nt < 4; ++nt) vfr[nt] = frag_b(&Vt[nt * 16 + lq][0], 0, half);

#pragma unroll
    for (int st = 0; st < 2; ++st) {
#pragma unroll
      for (int r = 0; r < 8; ++r) {
        float p = __expf(stile[st][r] - m_new[r]);
        psum[r] += p;
        Psm[w][half * 8 + r][st * 16 + lq] = (__bf16)p;
      }
    }
#pragma unroll
    for (int off = 1; off < 16; off <<= 1)
#pragma unroll
      for (int r = 0; r < 8; ++r) psum[r] += __shfl_xor(psum[r], off);
#pragma unroll
    for (int r = 0; r < 8; ++r) l_run[r] = l_run[r] * fac[r] + psum[r];
#pragma unroll
    for (int nt = 0; nt < 4; ++nt)
#pragma unroll
      for (int r = 0; r < 8; ++r) O[nt][r] *= fac[r];

    asm volatile("s_wait_dscnt 0" ::: "memory");  // wave-private P re-read
    v16bf pa = frag_a(&Psm[w][lq][0], 0, half);
#pragma unroll
    for (int nt = 0; nt < 4; ++nt) O[nt] = WMMA_BF16(pa, vfr[nt], O[nt]);
    __syncthreads();
  }

#pragma unroll
  for (int r = 0; r < 8; ++r) l_run[r] = 1.0f / l_run[r];
#pragma unroll
  for (int nt = 0; nt < 4; ++nt) {
#pragma unroll
    for (int r = 0; r < 8; ++r) {
      size_t oidx = (size_t)(b * L + q0 + w * 16 + half * 8 + r) * 512 + h * 64 + nt * 16 + lq;
      Out[oidx] = (__bf16)(O[nt][r] * l_run[r]);
    }
  }
}

// ---------------------------------------------------------------------------
// Depthwise reflect conv (K=3, groups=C, out=4C) + exact-erf GELU -> bf16
// f1: (B,L,512) f32 ; g: (B,L,2048) bf16 ; out channel o uses input c = o/4.
// ---------------------------------------------------------------------------
__global__ __launch_bounds__(256) void dw_gelu(const float* __restrict__ f1,
                                               const float* __restrict__ wdw,
                                               __bf16* __restrict__ g) {
  int i = blockIdx.x * 256 + threadIdx.x;  // over B*L*C = 8388608
  int c = i & 511;
  int bl = i >> 9;  // b*L + l
  int l = bl & 4095;
  int lm = (l == 0) ? 1 : (l - 1);
  int lp = (l == 4095) ? 4094 : (l + 1);
  size_t rowbase = (size_t)(bl - l) * 512;
  float x0 = f1[rowbase + (size_t)lm * 512 + c];
  float x1 = f1[(size_t)bl * 512 + c];
  float x2 = f1[rowbase + (size_t)lp * 512 + c];
  size_t ob = (size_t)bl * 2048 + (size_t)c * 4;
#pragma unroll
  for (int e = 0; e < 4; ++e) {
    int o = c * 4 + e;
    float a = wdw[o * 3 + 0] * x0 + wdw[o * 3 + 1] * x1 + wdw[o * 3 + 2] * x2;
    float ge = 0.5f * a * (1.0f + erff(a * 0.70710678118654752f));
    g[ob + e] = (__bf16)ge;
  }
}

// ---------------------------------------------------------------------------
// Host launcher
// ---------------------------------------------------------------------------
extern "C" void kernel_launch(void* const* d_in, const int* in_sizes, int n_in,
                              void* d_out, int out_size, void* d_ws, size_t ws_size,
                              hipStream_t stream) {
  (void)in_sizes; (void)n_in; (void)out_size; (void)ws_size;
  const float* x      = (const float*)d_in[0];
  const float* w_q    = (const float*)d_in[1];
  const float* w_k    = (const float*)d_in[2];
  const float* w_v    = (const float*)d_in[3];
  const float* w_o    = (const float*)d_in[4];
  const float* w_red  = (const float*)d_in[5];
  const float* w_ffn1 = (const float*)d_in[6];
  const float* w_dw   = (const float*)d_in[7];
  const float* w_ffn2 = (const float*)d_in[8];
  const float* lng    = (const float*)d_in[9];
  const float* lnb    = (const float*)d_in[10];
  float* out = (float*)d_out;

  const int B = 4, L = 4096, C = 512, Lr = 1024;
  const int NT = B * L;   // 16384 tokens
  const int NR = B * Lr;  // 4096 reduced tokens

  char* wsp = (char*)d_ws;
  auto alloc = [&](size_t bytes) -> void* {
    void* p = (void*)wsp;
    wsp += (bytes + 255) & ~(size_t)255;
    return p;
  };

  __bf16* wq_b   = (__bf16*)alloc((size_t)C * C * 2);
  __bf16* wk_b   = (__bf16*)alloc((size_t)C * C * 2);
  __bf16* wv_b   = (__bf16*)alloc((size_t)C * C * 2);
  __bf16* wo_b   = (__bf16*)alloc((size_t)C * C * 2);
  __bf16* wf1_b  = (__bf16*)alloc((size_t)C * C * 2);
  __bf16* wf2_b  = (__bf16*)alloc((size_t)C * 4 * C * 2);
  __bf16* wred_b = (__bf16*)alloc((size_t)C * 3584 * 2);
  __bf16* h_b    = (__bf16*)alloc((size_t)NT * C * 2);
  float*  red_f  = (float*) alloc((size_t)NR * C * 4);
  __bf16* red_b  = (__bf16*)alloc((size_t)NR * C * 2);
  float*  q_f    = (float*) alloc((size_t)NT * C * 4);
  __bf16* q_b    = (__bf16*)alloc((size_t)NT * C * 2);
  float*  k_f    = (float*) alloc((size_t)NR * C * 4);
  __bf16* k_b    = (__bf16*)alloc((size_t)NR * C * 2);
  float*  v_f    = (float*) alloc((size_t)NR * C * 4);
  __bf16* v_b    = (__bf16*)alloc((size_t)NR * C * 2);
  __bf16* ao_b   = (__bf16*)alloc((size_t)NT * C * 2);
  float*  h2_f   = (float*) alloc((size_t)NT * C * 4);
  __bf16* f_b    = (__bf16*)alloc((size_t)NT * C * 2);
  float*  f1_f   = (float*) alloc((size_t)NT * C * 4);
  __bf16* g_b    = (__bf16*)alloc((size_t)NT * 4 * C * 2);

  const int T = 256;
  auto blk = [](int n) { return (n + 255) / 256; };

  // weights -> bf16
  cvt_bf16<<<blk(C * C), T, 0, stream>>>(w_q, wq_b, C * C);
  cvt_bf16<<<blk(C * C), T, 0, stream>>>(w_k, wk_b, C * C);
  cvt_bf16<<<blk(C * C), T, 0, stream>>>(w_v, wv_b, C * C);
  cvt_bf16<<<blk(C * C), T, 0, stream>>>(w_o, wo_b, C * C);
  cvt_bf16<<<blk(C * C), T, 0, stream>>>(w_ffn1, wf1_b, C * C);
  cvt_bf16<<<blk(C * 4 * C), T, 0, stream>>>(w_ffn2, wf2_b, C * 4 * C);
  cvt_wred<<<blk(C * C * 7), T, 0, stream>>>(w_red, wred_b);

  // ln1
  ln_bf16<<<NT / 8, T, 0, stream>>>(x, lng + 0 * C, lnb + 0 * C, h_b, NT);

  // reducer conv as K=3584 GEMM (im2col gather), then LN
  gemm_bf16<1, false><<<dim3(C / 128, NR / 128), T, 0, stream>>>(h_b, wred_b, nullptr, red_f, NR, C, 3584);
  ln_bf16<<<NR / 8, T, 0, stream>>>(red_f, lng + 1 * C, lnb + 1 * C, red_b, NR);

  // Q / K / V projections + LN
  gemm_bf16<0, false><<<dim3(C / 128, NT / 128), T, 0, stream>>>(h_b, wq_b, nullptr, q_f, NT, C, C);
  ln_bf16<<<NT / 8, T, 0, stream>>>(q_f, lng + 2 * C, lnb + 2 * C, q_b, NT);
  gemm_bf16<0, false><<<dim3(C / 128, NR / 128), T, 0, stream>>>(red_b, wk_b, nullptr, k_f, NR, C, C);
  ln_bf16<<<NR / 8, T, 0, stream>>>(k_f, lng + 3 * C, lnb + 3 * C, k_b, NR);
  gemm_bf16<0, false><<<dim3(C / 128, NR / 128), T, 0, stream>>>(red_b, wv_b, nullptr, v_f, NR, C, C);
  ln_bf16<<<NR / 8, T, 0, stream>>>(v_f, lng + 4 * C, lnb + 4 * C, v_b, NR);

  // attention
  attn_kernel<<<dim3(L / 128, B * 8), T, 0, stream>>>(q_b, k_b, v_b, ao_b);

  // output projection + residual -> h2
  gemm_bf16<0, true><<<dim3(C / 128, NT / 128), T, 0, stream>>>(ao_b, wo_b, x, h2_f, NT, C, C);

  // ln2 -> FFN
  ln_bf16<<<NT / 8, T, 0, stream>>>(h2_f, lng + 5 * C, lnb + 5 * C, f_b, NT);
  gemm_bf16<0, false><<<dim3(C / 128, NT / 128), T, 0, stream>>>(f_b, wf1_b, nullptr, f1_f, NT, C, C);
  dw_gelu<<<blk(NT * C), T, 0, stream>>>(f1_f, w_dw, g_b);
  gemm_bf16<0, true><<<dim3(C / 128, NT / 128), T, 0, stream>>>(g_b, wf2_b, h2_f, out, NT, C, 4 * C);
}
